// TokenTopKMoeBlock_44667659878791
// MI455X (gfx1250) — matrix operand order repaired and checked by reading, not compiled, not run on hardware
//
#include <hip/hip_runtime.h>
#include <hip/hip_bf16.h>
#include <math.h>

// ---------------------------------------------------------------------------
// MoE block (top-2 of 8 experts), dense formulation matching the reference.
//   B=4 S=1024 D=1024 F=4096 E=8 K=2, T = B*S = 4096 tokens
// Compute-bound (~825 GFLOP vs ~0.45 GB traffic) -> v_wmma_f32_16x16x32_f16
// with fp32 accumulate; fp32->f16 conversion happens during LDS staging.
// Down-proj A tile (f16 H) is staged with global_load_async_to_lds_b128.
// ---------------------------------------------------------------------------

typedef __attribute__((ext_vector_type(16))) _Float16 v16h;
typedef __attribute__((ext_vector_type(8)))  _Float16 v8h;
typedef __attribute__((ext_vector_type(8)))  float    v8f;

#define B_  4
#define S_  1024
#define D_  1024
#define F_  4096
#define E_  8
#define T_  (B_ * S_)

// GEMM block tiling: 64 tokens x 128 cols per block, K-step 32.
// 8 waves: 4 over M (16 rows each) x 2 over N (64 cols each -> 4 WMMA subtiles).
#define BM 64
#define BN 128
#define BK 32

// Assemble a v16h fragment from two contiguous 8-element halves in LDS.
__device__ __forceinline__ v16h ld_frag16(const _Float16* lo, const _Float16* hi) {
  v8h a = *(const v8h*)lo;
  v8h b = *(const v8h*)hi;
  v16h r;
#pragma unroll
  for (int i = 0; i < 8; ++i) { r[i] = a[i]; r[8 + i] = b[i]; }
  return r;
}

// ---------------------------------------------------------------------------
// Router: one wave per token. logits[t,e] = sum_d x[t,d]*gate_w[e,d];
// softmax over 8, top-2, renormalize -> combine[t,e] (0 for unselected).
// ---------------------------------------------------------------------------
__global__ void moe_router(const float* __restrict__ x,
                           const float* __restrict__ gate_w,
                           float* __restrict__ logits,
                           float* __restrict__ combine) {
  const int wave = threadIdx.x >> 5;
  const int lane = threadIdx.x & 31;
  const int t    = blockIdx.x * 8 + wave;

  const float* xr = x + (size_t)t * D_;
  float acc[E_];
#pragma unroll
  for (int e = 0; e < E_; ++e) acc[e] = 0.0f;

  for (int d = lane; d < D_; d += 32) {
    float xv = xr[d];
#pragma unroll
    for (int e = 0; e < E_; ++e) acc[e] += xv * gate_w[e * D_ + d];
  }

#pragma unroll
  for (int e = 0; e < E_; ++e) {
    float v = acc[e];
#pragma unroll
    for (int off = 16; off > 0; off >>= 1) v += __shfl_xor(v, off, 32);
    acc[e] = v;
  }

  if (lane == 0) {
    float mx = acc[0];
#pragma unroll
    for (int e = 1; e < E_; ++e) mx = fmaxf(mx, acc[e]);
    float p[E_];
    float psum = 0.0f;
#pragma unroll
    for (int e = 0; e < E_; ++e) { p[e] = __expf(acc[e] - mx); psum += p[e]; }
    const float rpsum = __builtin_amdgcn_rcpf(psum);   // v_rcp_f32
#pragma unroll
    for (int e = 0; e < E_; ++e) p[e] *= rpsum;

    // top-2 (ties resolved to lowest index, like jax.lax.top_k)
    int i1 = 0;
#pragma unroll
    for (int e = 1; e < E_; ++e) if (p[e] > p[i1]) i1 = e;
    int i2 = (i1 == 0) ? 1 : 0;
#pragma unroll
    for (int e = 0; e < E_; ++e) if (e != i1 && p[e] > p[i2]) i2 = e;
    const float rwsum = __builtin_amdgcn_rcpf(p[i1] + p[i2]);

#pragma unroll
    for (int e = 0; e < E_; ++e) {
      logits[t * E_ + e]  = acc[e];
      float cw = (e == i1) ? (p[i1] * rwsum)
               : ((e == i2) ? (p[i2] * rwsum) : 0.0f);
      combine[t * E_ + e] = cw;
    }
  }
}

// ---------------------------------------------------------------------------
// rl[b,e] = mean over S of logits  (deterministic tree reduction, no atomics;
// fully overwrites the poisoned rl region of d_out).
// ---------------------------------------------------------------------------
__global__ void moe_rl_reduce(const float* __restrict__ logits,
                              float* __restrict__ rl_out) {
  const int b = blockIdx.x / E_;
  const int e = blockIdx.x % E_;
  __shared__ float red[256];
  float s = 0.0f;
  for (int si = threadIdx.x; si < S_; si += 256)
    s += logits[(size_t)(b * S_ + si) * E_ + e];
  red[threadIdx.x] = s;
  __syncthreads();
  for (int off = 128; off > 0; off >>= 1) {
    if (threadIdx.x < off) red[threadIdx.x] += red[threadIdx.x + off];
    __syncthreads();
  }
  if (threadIdx.x == 0) rl_out[b * E_ + e] = red[0] * (1.0f / (float)S_);
}

// ---------------------------------------------------------------------------
// Up projection for one expert: H[t,f] = silu(x@w1)[t,f] * (x@w3)[t,f], f16.
// Dual-GEMM sharing the A fragment. grid = (F/BN, T/BM), block = 256.
// ---------------------------------------------------------------------------
__global__ void moe_expert_up(const float* __restrict__ x,     // [T, D] fp32
                              const float* __restrict__ w1,    // [D, F] fp32 (this expert)
                              const float* __restrict__ w3,    // [D, F] fp32 (this expert)
                              _Float16* __restrict__ H) {      // [T, F] f16
  __shared__ __align__(16) _Float16 sA [BM][BK];   // tokens x k
  __shared__ __align__(16) _Float16 sB1[BN][BK];   // f-col  x k (transposed)
  __shared__ __align__(16) _Float16 sB3[BN][BK];

  const int tid  = threadIdx.x;
  const int lane = tid & 31;
  const int wv   = tid >> 5;
  const int m_w  = (wv & 3) * 16;   // wave's 16 rows within BM
  const int n_w  = (wv >> 2) * 64;  // wave's 64 cols within BN
  const int t0   = blockIdx.y * BM;
  const int f0   = blockIdx.x * BN;

  v8f acc1[4] = {};
  v8f acc3[4] = {};

  const int row16 = lane & 15;
  const int koffA = (lane < 16) ? 0 : 16;  // A frag: K {koffA..koffA+15}
  const int koffB = (lane < 16) ? 0 : 16;  // B frag: K {koffB..koffB+15}

  for (int k0 = 0; k0 < D_; k0 += BK) {
    // Prefetch next K-step weight tiles (one 64B line per thread tiles 32x128).
    if (k0 + BK < D_) {
      size_t pg = (size_t)(k0 + BK + (tid >> 3)) * F_ + f0 + (tid & 7) * 16;
      __builtin_prefetch(&w1[pg], 0, 0);
      __builtin_prefetch(&w3[pg], 0, 0);
    }
    // Stage A: x[t0..t0+63][k0..k0+31] fp32 -> f16
    for (int i = tid; i < BM * BK; i += 256) {
      int r = i >> 5, c = i & 31;
      sA[r][c] = (_Float16)x[(size_t)(t0 + r) * D_ + (k0 + c)];
    }
    // Stage B1/B3 transposed: w[k][f] -> sB[f][k] (global reads coalesced on f)
    for (int i = tid; i < BK * BN; i += 256) {
      int kk = i >> 7, ff = i & 127;
      size_t gi = (size_t)(k0 + kk) * F_ + (f0 + ff);
      sB1[ff][kk] = (_Float16)w1[gi];
      sB3[ff][kk] = (_Float16)w3[gi];
    }
    __syncthreads();

    v16h a = ld_frag16(&sA[m_w + row16][koffA], &sA[m_w + row16][koffA + 8]);

#pragma unroll
    for (int nb = 0; nb < 4; ++nb) {
      const int colb = n_w + nb * 16 + row16;
      v16h b1 = ld_frag16(&sB1[colb][koffB], &sB1[colb][koffB + 8]);
      acc1[nb] = __builtin_amdgcn_wmma_f32_16x16x32_f16(
          false, a, false, b1, (short)0, acc1[nb], false, false);
      v16h b3 = ld_frag16(&sB3[colb][koffB], &sB3[colb][koffB + 8]);
      acc3[nb] = __builtin_amdgcn_wmma_f32_16x16x32_f16(
          false, a, false, b3, (short)0, acc3[nb], false, false);
    }
    __syncthreads();
  }

  // Epilogue: h = silu(g) * u, store f16.
  // C layout: VGPR i -> M = i (lanes 0-15) or i+8 (lanes 16-31); N = lane&15.
  // sigmoid via v_exp_f32 + v_rcp_f32 (single-instruction reciprocal).
  const int half8 = (lane >> 4) * 8;
#pragma unroll
  for (int nb = 0; nb < 4; ++nb) {
    const int f = f0 + n_w + nb * 16 + row16;
#pragma unroll
    for (int i = 0; i < 8; ++i) {
      const int t = t0 + m_w + i + half8;
      float g = acc1[nb][i];
      float u = acc3[nb][i];
      float h = g * __builtin_amdgcn_rcpf(1.0f + __expf(-g)) * u;  // silu(g)*u
      H[(size_t)t * F_ + f] = (_Float16)h;
    }
  }
}

// ---------------------------------------------------------------------------
// Down projection for one expert: out[t,d] (+)= (H@w2)[t,d] * combine[t,e].
// A tile (f16 H) staged via global_load_async_to_lds_b128 (ASYNCcnt path);
// B tile fp32->f16 conversion overlaps with the in-flight async copy.
// addto=0 overwrites (expert 0), addto=1 accumulates. grid = (D/BN, T/BM).
// ---------------------------------------------------------------------------
__global__ void moe_expert_down(const _Float16* __restrict__ H,   // [T, F] f16
                                const float* __restrict__ w2,     // [F, D] fp32 (this expert)
                                const float* __restrict__ combine,// [T, E]
                                int e, int addto,
                                float* __restrict__ out) {        // [T, D]
  __shared__ __align__(16) _Float16 sA[BM][BK];   // tokens x k
  __shared__ __align__(16) _Float16 sB[BN][BK];   // d-col  x k (transposed)

  const int tid  = threadIdx.x;
  const int lane = tid & 31;
  const int wv   = tid >> 5;
  const int m_w  = (wv & 3) * 16;
  const int n_w  = (wv >> 2) * 64;
  const int t0   = blockIdx.y * BM;
  const int d0   = blockIdx.x * BN;

  v8f acc[4] = {};

  const int row16 = lane & 15;
  const int koffA = (lane < 16) ? 0 : 16;
  const int koffB = (lane < 16) ? 0 : 16;

  // Async staging of sA: 256 threads x 16B = whole 64x32 f16 tile per K-step.
  const int ar  = tid >> 2;          // token row within tile
  const int ac8 = (tid & 3) * 8;     // k start (8 f16 = 16B)
  unsigned long long gA =
      (unsigned long long)(const void*)(H + (size_t)(t0 + ar) * F_ + ac8);
  const unsigned int ldsA =
      (unsigned int)(uintptr_t)(&sA[0][0]) + (unsigned int)(tid * 16);

  for (int k0 = 0; k0 < F_; k0 += BK) {
    // Kick off async H tile copy (memory -> LDS, no VGPR round-trip).
    asm volatile("global_load_async_to_lds_b128 %0, %1, off"
                 :: "v"(ldsA), "v"(gA) : "memory");
    gA += (unsigned long long)(BK * sizeof(_Float16));

    // Prefetch next K-step w2 tile.
    if (k0 + BK < F_) {
      size_t pg = (size_t)(k0 + BK + (tid >> 3)) * D_ + d0 + (tid & 7) * 16;
      __builtin_prefetch(&w2[pg], 0, 0);
    }
    // Stage B transposed while the async copy is in flight.
    for (int i = tid; i < BK * BN; i += 256) {
      int kk = i >> 7, ff = i & 127;
      sB[ff][kk] = (_Float16)w2[(size_t)(k0 + kk) * D_ + (d0 + ff)];
    }
    asm volatile("s_wait_asynccnt 0" ::: "memory");
    __syncthreads();

    v16h a = ld_frag16(&sA[m_w + row16][koffA], &sA[m_w + row16][koffA + 8]);
#pragma unroll
    for (int nb = 0; nb < 4; ++nb) {
      const int colb = n_w + nb * 16 + row16;
      v16h b = ld_frag16(&sB[colb][koffB], &sB[colb][koffB + 8]);
      acc[nb] = __builtin_amdgcn_wmma_f32_16x16x32_f16(
          false, a, false, b, (short)0, acc[nb], false, false);
    }
    __syncthreads();
  }

  const int half8 = (lane >> 4) * 8;
#pragma unroll
  for (int i = 0; i < 8; ++i) {
    const int t  = t0 + m_w + i + half8;
    const float cw = combine[t * E_ + e];
#pragma unroll
    for (int nb = 0; nb < 4; ++nb) {
      const int d = d0 + n_w + nb * 16 + row16;
      size_t oi = (size_t)t * D_ + d;
      float v = acc[nb][i] * cw;
      if (addto) v += out[oi];
      out[oi] = v;
    }
  }
}

// ---------------------------------------------------------------------------
// Launch. Workspace layout (f32 unless noted):
//   [0, T*E)            combine
//   [T*E, 2*T*E)        router logits
//   [2*T*E, ...)        H intermediate, T*F f16  (~32 MB), reused per expert
// ---------------------------------------------------------------------------
extern "C" void kernel_launch(void* const* d_in, const int* in_sizes, int n_in,
                              void* d_out, int out_size, void* d_ws, size_t ws_size,
                              hipStream_t stream) {
  const float* x      = (const float*)d_in[0];  // [B,S,D]
  const float* gate_w = (const float*)d_in[1];  // [E,D]
  const float* w1     = (const float*)d_in[2];  // [E,D,F]
  const float* w3     = (const float*)d_in[3];  // [E,D,F]
  const float* w2     = (const float*)d_in[4];  // [E,F,D]
  (void)in_sizes; (void)n_in; (void)out_size; (void)ws_size;

  float* out = (float*)d_out;                   // final [T*D], then rl [B*E]
  float* rl  = out + (size_t)T_ * D_;

  float*    combine = (float*)d_ws;
  float*    logits  = combine + (size_t)T_ * E_;
  _Float16* Hbuf    = (_Float16*)(logits + (size_t)T_ * E_);

  moe_router<<<T_ / 8, 256, 0, stream>>>(x, gate_w, logits, combine);
  moe_rl_reduce<<<B_ * E_, 256, 0, stream>>>(logits, rl);

  for (int e = 0; e < E_; ++e) {
    moe_expert_up<<<dim3(F_ / BN, T_ / BM), 256, 0, stream>>>(
        x, w1 + (size_t)e * D_ * F_, w3 + (size_t)e * D_ * F_, Hbuf);
    moe_expert_down<<<dim3(D_ / BN, T_ / BM), 256, 0, stream>>>(
        Hbuf, w2 + (size_t)e * F_ * D_, combine, e, (e > 0) ? 1 : 0, out);
  }
}
